// GatedDeltaNet_32074815767386
// MI455X (gfx1250) — matrix operand order repaired and checked by reading, not compiled, not run on hardware
//
#include <hip/hip_runtime.h>
#include <cstdint>
#include <cstddef>

// ---------------------------------------------------------------------------
// Types for CDNA5 WMMA (wave32, 16x16x32 bf16 -> f32)
// ---------------------------------------------------------------------------
typedef __attribute__((ext_vector_type(16))) __bf16  v16bf;
typedef __attribute__((ext_vector_type(8)))  float   v8f;
typedef __attribute__((ext_vector_type(4)))  unsigned int u32x4;

union FragBF { v16bf v; u32x4 q[2]; };

#if __has_builtin(__builtin_amdgcn_sched_group_barrier)
#define SCHED_GROUP(mask, size, id) __builtin_amdgcn_sched_group_barrier(mask, size, id)
#else
#define SCHED_GROUP(mask, size, id)
#endif

__device__ __forceinline__ uint16_t f2bf(float x) {
    uint32_t u = __float_as_uint(x);
    uint32_t r = u + 0x7FFFu + ((u >> 16) & 1u);   // round-to-nearest-even
    return (uint16_t)(r >> 16);
}
__device__ __forceinline__ float bf2f(uint16_t b) {
    return __uint_as_float(((uint32_t)b) << 16);
}

// ---------------------------------------------------------------------------
// CDNA5 async global->LDS copy (ASYNCcnt-tracked, no VGPR data movement)
// ---------------------------------------------------------------------------
__device__ __forceinline__ void async_load_b128(uint32_t lds_byte_addr, const void* gaddr) {
    asm volatile("global_load_async_to_lds_b128 %0, %1, off"
                 :: "v"(lds_byte_addr), "v"(gaddr)
                 : "memory");
}
__device__ __forceinline__ void wait_async0() {
#if __has_builtin(__builtin_amdgcn_s_wait_asynccnt)
    __builtin_amdgcn_s_wait_asynccnt(0);
#else
    asm volatile("s_wait_asynccnt 0x0" ::: "memory");
#endif
}

// A fragment (16x32 bf16, row-major source, stride lda elements).
__device__ __forceinline__ v16bf fragA(const uint16_t* A, int lda, int sm, int k0, int lane) {
    FragBF f;
    const uint16_t* p = A + (size_t)(sm + (lane & 15)) * lda + k0 + ((lane >> 4) << 4);
    f.q[0] = *(const u32x4*)p;
    f.q[1] = *(const u32x4*)(p + 8);
    return f.v;
}
// B fragment from Bt[N][K] (row-major along K): lane L holds column sn+(L&15),
// 16 consecutive K starting at k0 + 16*(L>>4)  -> one contiguous 32-byte read.
__device__ __forceinline__ v16bf fragB(const uint16_t* Bt, int ldb, int sn, int k0, int lane) {
    FragBF f;
    const uint16_t* p = Bt + (size_t)(sn + (lane & 15)) * ldb + k0 + ((lane >> 4) << 4);
    f.q[0] = *(const u32x4*)p;
    f.q[1] = *(const u32x4*)(p + 8);
    return f.v;
}

// ---------------------------------------------------------------------------
// LDS tile matmul helper:  C(MtxNt) (+)= A(MtxKt) * Bt(NtxKt)^T, bf16 in, f32 out
// 8 waves split the 16x16 subtiles.  MODE: 0 assign, 1 add, 2 subtract.
// ---------------------------------------------------------------------------
template <int MODE>
__device__ __forceinline__ void lds_mm_f32(const uint16_t* A, int lda,
                                           const uint16_t* Bt, int ldb,
                                           float* out, int ldo,
                                           int Mt, int Nt, int Kt,
                                           int wave, int lane) {
    const int nm = Mt >> 4, nn = Nt >> 4;
    for (int s = wave; s < nm * nn; s += 8) {
        const int sm = (s % nm) << 4, sn = (s / nm) << 4;
        v8f acc = {};
        for (int k0 = 0; k0 < Kt; k0 += 32) {
            v16bf a = fragA(A, lda, sm, k0, lane);
            v16bf b = fragB(Bt, ldb, sn, k0, lane);
            acc = __builtin_amdgcn_wmma_f32_16x16x32_bf16(false, a, false, b,
                                                          (short)0, acc, false, false);
            SCHED_GROUP(0x100, 4, 0);   // 4 ds reads
            SCHED_GROUP(0x008, 1, 0);   // then the wmma
        }
        const int m0 = sm + ((lane >> 4) << 3);
        const int n  = sn + (lane & 15);
#pragma unroll
        for (int r = 0; r < 8; ++r) {
            float* p = out + (size_t)(m0 + r) * ldo + n;
            if (MODE == 0)      *p = acc[r];
            else if (MODE == 1) *p += acc[r];
            else                *p -= acc[r];
        }
    }
}

// Same but writes bf16 (assign only)
__device__ __forceinline__ void lds_mm_bf16(const uint16_t* A, int lda,
                                            const uint16_t* Bt, int ldb,
                                            uint16_t* out, int ldo,
                                            int Mt, int Nt, int Kt,
                                            int wave, int lane) {
    const int nm = Mt >> 4, nn = Nt >> 4;
    for (int s = wave; s < nm * nn; s += 8) {
        const int sm = (s % nm) << 4, sn = (s / nm) << 4;
        v8f acc = {};
        for (int k0 = 0; k0 < Kt; k0 += 32) {
            v16bf a = fragA(A, lda, sm, k0, lane);
            v16bf b = fragB(Bt, ldb, sn, k0, lane);
            acc = __builtin_amdgcn_wmma_f32_16x16x32_bf16(false, a, false, b,
                                                          (short)0, acc, false, false);
            SCHED_GROUP(0x100, 4, 0);
            SCHED_GROUP(0x008, 1, 0);
        }
        const int m0 = sm + ((lane >> 4) << 3);
        const int n  = sn + (lane & 15);
#pragma unroll
        for (int r = 0; r < 8; ++r)
            out[(size_t)(m0 + r) * ldo + n] = f2bf(acc[r]);
    }
}

// ---------------------------------------------------------------------------
// Kernel 0: f32 -> bf16 flat convert
// ---------------------------------------------------------------------------
__global__ void cvt_bf16_kernel(const float* __restrict__ in, uint16_t* __restrict__ out,
                                size_t n) {
    size_t i = (size_t)blockIdx.x * 256 + threadIdx.x;
    if (i < n) out[i] = f2bf(in[i]);
}

// Kernel 1: transpose+convert:  in[Kd][Nd] f32 -> out[Nd][Kd] bf16
__global__ void transpose_bf16_kernel(const float* __restrict__ in, uint16_t* __restrict__ out,
                                      int Kd, int Nd) {
    size_t i = (size_t)blockIdx.x * 256 + threadIdx.x;
    size_t tot = (size_t)Kd * Nd;
    if (i < tot) {
        int n = (int)(i / (size_t)Kd);
        int k = (int)(i % (size_t)Kd);
        out[i] = f2bf(in[(size_t)k * Nd + n]);
    }
}

// ---------------------------------------------------------------------------
// Kernel 2: big WMMA GEMM.  C[M][N] f32 = A[M][K] bf16 (row major) x Bt[N][K] bf16.
// Block = 8 waves in a 4(M) x 2(N) grid; block tile 128x128; each wave owns a
// 32(M) x 64(N) tile = 2x4 subtiles -> 12 b128 loads feed 8 WMMAs per K-step.
// sched_group_barrier pins "loads first, then WMMAs" so partial loadcnt waits
// let matrix ops overlap outstanding loads.
// ---------------------------------------------------------------------------
__global__ __launch_bounds__(256) void gemm_bf16_nt(const uint16_t* __restrict__ A,
                                                    const uint16_t* __restrict__ Bt,
                                                    float* __restrict__ C,
                                                    int M, int N, int K) {
    const int lane = threadIdx.x & 31, wave = threadIdx.x >> 5;
    const int tM = blockIdx.y * 128 + (wave & 3) * 32;
    const int tN = blockIdx.x * 128 + (wave >> 2) * 64;
    v8f acc[2][4];
#pragma unroll
    for (int i = 0; i < 2; ++i)
#pragma unroll
        for (int j = 0; j < 4; ++j) acc[i][j] = {};

    for (int k0 = 0; k0 < K; k0 += 32) {
        FragBF fa[2], fb[4];
#pragma unroll
        for (int i = 0; i < 2; ++i) {
            const uint16_t* pa = A + (size_t)(tM + i * 16 + (lane & 15)) * K +
                                 k0 + ((lane >> 4) << 4);
            fa[i].q[0] = *(const u32x4*)pa;
            fa[i].q[1] = *(const u32x4*)(pa + 8);
        }
#pragma unroll
        for (int j = 0; j < 4; ++j) {
            const uint16_t* pb = Bt + (size_t)(tN + j * 16 + (lane & 15)) * K +
                                 k0 + ((lane >> 4) << 4);
            fb[j].q[0] = *(const u32x4*)pb;
            fb[j].q[1] = *(const u32x4*)(pb + 8);
        }
        if (k0 + 32 < K)
            __builtin_prefetch(A + (size_t)(tM + (lane & 15)) * K + k0 + 32, 0, 1);
#pragma unroll
        for (int i = 0; i < 2; ++i)
#pragma unroll
            for (int j = 0; j < 4; ++j)
                acc[i][j] = __builtin_amdgcn_wmma_f32_16x16x32_bf16(
                    false, fa[i].v, false, fb[j].v, (short)0, acc[i][j], false, false);
        SCHED_GROUP(0x020, 13, 0);   // all vmem reads (12 frags + prefetch) first
        SCHED_GROUP(0x008, 8, 0);    // then the 8 wmma
    }
    const int nc = lane & 15;
#pragma unroll
    for (int i = 0; i < 2; ++i) {
        const int m0 = tM + i * 16 + ((lane >> 4) << 3);
#pragma unroll
        for (int j = 0; j < 4; ++j)
#pragma unroll
            for (int r = 0; r < 8; ++r)
                C[(size_t)(m0 + r) * N + tN + j * 16 + nc] = acc[i][j][r];
    }
}

// ---------------------------------------------------------------------------
// Kernel 3: ba = hidden @ W_ba ; beta = sigmoid(b) ; g = -exp(A_log)*softplus(a+dt_bias)
// ---------------------------------------------------------------------------
__global__ __launch_bounds__(256) void ba_gate_kernel(const float* __restrict__ hid,
                                                      const float* __restrict__ Wba,
                                                      const float* __restrict__ dt_bias,
                                                      const float* __restrict__ A_log,
                                                      float* __restrict__ beta,
                                                      float* __restrict__ g) {
    __shared__ float row[2048];
    const int t = blockIdx.x, tid = threadIdx.x;
    for (int i = tid; i < 2048; i += 256) row[i] = hid[(size_t)t * 2048 + i];
    __syncthreads();
    if (tid < 64) {
        float s = 0.f;
        for (int d = 0; d < 2048; ++d) s += row[d] * Wba[(size_t)d * 64 + tid];
        const int hkk = tid >> 2, r = tid & 3;
        const int hv = hkk * 2 + (r & 1);
        if (r < 2) {
            beta[(size_t)t * 32 + hv] = 1.f / (1.f + expf(-s));
        } else {
            float x = s + dt_bias[hv];
            float sp = (x > 20.f) ? x : log1pf(expf(x));
            g[(size_t)t * 32 + hv] = -expf(A_log[hv]) * sp;
        }
    }
}

// ---------------------------------------------------------------------------
// Kernel 4: causal conv(K=4) + SiLU + per-head L2 norm -> q,k bf16, v f32
// ---------------------------------------------------------------------------
__device__ __forceinline__ int mixed_col(int c) {
    if (c < 2048)  return (c >> 7) * 768 + (c & 127);                 // q
    if (c < 4096) { int c2 = c - 2048; return (c2 >> 7) * 768 + 128 + (c2 & 127); } // k
    int c3 = c - 4096; return (c3 >> 8) * 768 + 256 + (c3 & 255);     // v
}

__global__ __launch_bounds__(256) void conv_silu_norm_kernel(const float* __restrict__ qkvz,
                                                             const float* __restrict__ convw,
                                                             uint16_t* __restrict__ qg,
                                                             uint16_t* __restrict__ kg,
                                                             float* __restrict__ vg) {
    __shared__ float buf[8192];
    const int t = blockIdx.x, tid = threadIdx.x;
    for (int c = tid; c < 8192; c += 256) {
        const int col = mixed_col(c);
        float acc = 0.f;
#pragma unroll
        for (int j = 0; j < 4; ++j) {
            const int tp = t + j - 3;
            if (tp >= 0) acc += convw[c * 4 + j] * qkvz[(size_t)tp * 12288 + col];
        }
        buf[c] = acc / (1.f + expf(-acc));   // SiLU
    }
    __syncthreads();
    const int lane = tid & 31, wave = tid >> 5;
    for (int i = 0; i < 4; ++i) {
        const int tau = wave * 4 + i;        // 0..15 = q heads, 16..31 = k heads
        const int base = tau * 128;
        float v0[4]; float ss = 0.f;
#pragma unroll
        for (int jj = 0; jj < 4; ++jj) { v0[jj] = buf[base + lane + 32 * jj]; ss += v0[jj] * v0[jj]; }
#pragma unroll
        for (int m = 16; m >= 1; m >>= 1) ss += __shfl_xor(ss, m, 32);
        const float inv = rsqrtf(ss + 1e-6f);
        uint16_t* dst = (tau < 16) ? (qg + (size_t)t * 2048 + base)
                                   : (kg + (size_t)t * 2048 + (base - 2048));
#pragma unroll
        for (int jj = 0; jj < 4; ++jj) dst[lane + 32 * jj] = f2bf(v0[jj] * inv);
    }
    for (int i = tid; i < 4096; i += 256) vg[(size_t)t * 4096 + i] = buf[4096 + i];
}

// ---------------------------------------------------------------------------
// Kernel 5: chunked gated delta rule.  One block (8 waves) per value-head,
// sequential over 64 chunks of 64 timesteps.  State S^T[dv][dk] f32 in LDS.
// All matmuls via WMMA on LDS bf16 tiles; q/k chunk tiles are staged with
// CDNA5 async global->LDS (ASYNCcnt) copies.
// ---------------------------------------------------------------------------
#define DELTA_SMEM 295424
#define Q_OFF  98304
#define K_OFF  114688

__global__ __launch_bounds__(256) void delta_rule_kernel(const uint16_t* __restrict__ qgm,
                                                         const uint16_t* __restrict__ kgm,
                                                         const float* __restrict__ vgm,
                                                         const float* __restrict__ ggm,
                                                         const float* __restrict__ bgm,
                                                         float* __restrict__ og) {
    extern __shared__ char smem[];
    float*    St   = (float*)(smem + 0);          // [128][128] f32  (S^T: dv x dk)
    uint16_t* Stb  = (uint16_t*)(smem + 65536);   // [128][128] bf16
    uint16_t* Q    = (uint16_t*)(smem + Q_OFF);   // [64][128]  q -> qd (in place)
    uint16_t* Kb   = (uint16_t*)(smem + K_OFF);   // [64][128]  raw k
    float*    Mf   = (float*)(smem + 131072);     // [64][64]
    float*    Tf   = (float*)(smem + 147456);     // [64][64]  Tinv f32
    uint16_t* Ab   = (uint16_t*)(smem + 163840);  // [64][64]
    uint16_t* Tb   = (uint16_t*)(smem + 172032);  // [64][64]
    uint16_t* bvT  = (uint16_t*)(smem + 180224);  // [128][64]
    uint16_t* bgkT = (uint16_t*)(smem + 196608);  // [128][64]
    uint16_t* Wb   = (uint16_t*)(smem + 212992);  // [64][128]
    float*    Uf   = (float*)(smem + 229376);     // [64][128] u then o
    uint16_t* uTb  = (uint16_t*)(smem + 262144);  // [128][64]
    uint16_t* kdT  = (uint16_t*)(smem + 278528);  // [128][64]
    float*    Gc   = (float*)(smem + 294912);     // [64] cumsum(g)
    float*    Bc   = (float*)(smem + 295168);     // [64] beta

    const int tid = threadIdx.x, lane = tid & 31, wave = tid >> 5;
    const int h = blockIdx.x, hk = h >> 1;
    const float scale = 0.0883883476483184f;      // 128^-0.5

    for (int i = tid; i < 128 * 128; i += 256) St[i] = 0.f;
    __syncthreads();

    for (int c = 0; c < 64; ++c) {
        const int t0 = c * 64;
        // ---- async-stage chunk q,k (shared kv head) straight into LDS ----
        for (int i = tid; i < 1024; i += 256) {
            const int tt = i >> 4, d8 = (i & 15) << 3;
            const uint32_t dst = (uint32_t)((tt << 7) + d8) * 2;
            async_load_b128(Q_OFF + dst, qgm + (size_t)(t0 + tt) * 2048 + hk * 128 + d8);
            async_load_b128(K_OFF + dst, kgm + (size_t)(t0 + tt) * 2048 + hk * 128 + d8);
        }
        if (tid == 0) {
            float s = 0.f;
            for (int tt = 0; tt < 64; ++tt) { s += ggm[(size_t)(t0 + tt) * 32 + h]; Gc[tt] = s; }
        }
        for (int i = tid; i < 64; i += 256) Bc[i] = bgm[(size_t)(t0 + i) * 32 + h];
        wait_async0();
        __syncthreads();

        // ---- KK = k k^T -> Mf ; QK = q k^T -> Tf ----
        lds_mm_f32<0>(Kb, 128, Kb, 128, Mf, 64, 64, 64, 128, wave, lane);
        lds_mm_f32<0>(Q, 128, Kb, 128, Tf, 64, 64, 64, 128, wave, lane);
        __syncthreads();

        // ---- masks / gating factors ----
        for (int i = tid; i < 64 * 64; i += 256) {
            const int tt = i >> 6, s = i & 63;
            const float e = (tt >= s) ? expf(Gc[tt] - Gc[s]) : 0.f;
            Ab[i] = f2bf((tt >= s) ? scale * e * Tf[i] : 0.f);
            Mf[i] = (tt > s) ? Bc[tt] * e * Mf[i] : 0.f;
        }
        __syncthreads();
        for (int i = tid; i < 64 * 64; i += 256) Tf[i] = ((i >> 6) == (i & 63)) ? 1.f : 0.f;
        __syncthreads();

        // ---- Tinv = (I + M)^-1 via forward substitution ----
        for (int i2 = 1; i2 < 64; ++i2) {
            if (tid < 64) {
                float s = 0.f;
                for (int l = 0; l < i2; ++l) s += Mf[i2 * 64 + l] * Tf[l * 64 + tid];
                Tf[i2 * 64 + tid] = ((i2 == tid) ? 1.f : 0.f) - s;
            }
            __syncthreads();
        }
        for (int i = tid; i < 64 * 64; i += 256) Tb[i] = f2bf(Tf[i]);

        // ---- bvT, bgkT, Stb ----
        for (int i = tid; i < 64 * 128; i += 256) {
            const int tt = i >> 7, d = i & 127;
            bvT[d * 64 + tt]  = f2bf(Bc[tt] * vgm[(size_t)(t0 + tt) * 4096 + h * 128 + d]);
            bgkT[d * 64 + tt] = f2bf(Bc[tt] * expf(Gc[tt]) * bf2f(Kb[i]));
        }
        for (int i = tid; i < 128 * 128; i += 256) Stb[i] = f2bf(St[i]);
        __syncthreads();

        // ---- u0 = Tinv @ (beta*v) ; w = Tinv @ (beta*exp(G)*k) ----
        lds_mm_f32<0>(Tb, 64, bvT, 64, Uf, 128, 64, 128, 64, wave, lane);
        lds_mm_bf16(Tb, 64, bgkT, 64, Wb, 128, 64, 128, 64, wave, lane);
        __syncthreads();

        // ---- u = u0 - w @ S ----
        lds_mm_f32<2>(Wb, 128, Stb, 128, Uf, 128, 64, 128, 128, wave, lane);
        __syncthreads();

        // ---- uT, kdT, qd (in place) ----
        for (int i = tid; i < 64 * 128; i += 256) {
            const int tt = i >> 7, d = i & 127;
            uTb[d * 64 + tt] = f2bf(Uf[i]);
            kdT[d * 64 + tt] = f2bf(expf(Gc[63] - Gc[tt]) * bf2f(Kb[i]));
            Q[i] = f2bf(scale * expf(Gc[tt]) * bf2f(Q[i]));
        }
        __syncthreads();

        // ---- o = A @ u + qd @ S ----
        lds_mm_f32<0>(Ab, 64, uTb, 64, Uf, 128, 64, 128, 64, wave, lane);
        __syncthreads();
        lds_mm_f32<1>(Q, 128, Stb, 128, Uf, 128, 64, 128, 128, wave, lane);
        __syncthreads();
        for (int i = tid; i < 64 * 128; i += 256) {
            const int tt = i >> 7, d = i & 127;
            og[((size_t)(t0 + tt) * 32 + h) * 128 + d] = Uf[i];
        }

        // ---- S = exp(gC)*S + kd^T @ u  (as S^T += u^T @ kd) ----
        const float dec = expf(Gc[63]);
        for (int i = tid; i < 128 * 128; i += 256) St[i] *= dec;
        __syncthreads();
        lds_mm_f32<1>(uTb, 64, kdT, 64, St, 128, 128, 128, 64, wave, lane);
        __syncthreads();
    }
}

// ---------------------------------------------------------------------------
// Kernel 6: xg = RMSNorm(o * silu(z)) * norm_weight  -> bf16 [T][4096]
// ---------------------------------------------------------------------------
__global__ __launch_bounds__(256) void gate_norm_kernel(const float* __restrict__ o,
                                                        const float* __restrict__ qkvz,
                                                        const float* __restrict__ normw,
                                                        uint16_t* __restrict__ xg) {
    __shared__ float buf[4096];
    const int t = blockIdx.x, tid = threadIdx.x;
    for (int i = tid; i < 4096; i += 256) {
        const int h = i >> 7, d = i & 127;
        const int zcol = (h >> 1) * 768 + 512 + (h & 1) * 128 + d;
        const float z = qkvz[(size_t)t * 12288 + zcol];
        buf[i] = o[((size_t)t * 32 + h) * 128 + d] * (z / (1.f + expf(-z)));
    }
    __syncthreads();
    const int lane = tid & 31, wave = tid >> 5;
    for (int i = 0; i < 4; ++i) {
        const int h = wave * 4 + i, base = h * 128;
        float v0[4]; float ss = 0.f;
#pragma unroll
        for (int jj = 0; jj < 4; ++jj) { v0[jj] = buf[base + lane + 32 * jj]; ss += v0[jj] * v0[jj]; }
#pragma unroll
        for (int m = 16; m >= 1; m >>= 1) ss += __shfl_xor(ss, m, 32);
        const float inv = rsqrtf(ss * (1.f / 128.f) + 1e-6f);
#pragma unroll
        for (int jj = 0; jj < 4; ++jj) {
            const int d = lane + 32 * jj;
            xg[(size_t)t * 4096 + base + d] = f2bf(v0[jj] * inv * normw[d]);
        }
    }
}

// ---------------------------------------------------------------------------
// Host launcher
// ---------------------------------------------------------------------------
extern "C" void kernel_launch(void* const* d_in, const int* in_sizes, int n_in,
                              void* d_out, int out_size, void* d_ws, size_t ws_size,
                              hipStream_t stream) {
    (void)in_sizes; (void)n_in; (void)out_size; (void)ws_size;
    const float* hidden  = (const float*)d_in[0];
    const float* W_qkvz  = (const float*)d_in[1];
    const float* W_ba    = (const float*)d_in[2];
    const float* conv_w  = (const float*)d_in[3];
    const float* dt_bias = (const float*)d_in[4];
    const float* A_log   = (const float*)d_in[5];
    const float* norm_w  = (const float*)d_in[6];
    const float* W_out   = (const float*)d_in[7];
    float* out = (float*)d_out;

    // workspace partition
    char* ws = (char*)d_ws;
    size_t off = 0;
    auto take = [&](size_t bytes) -> void* {
        void* p = ws + off;
        off += (bytes + 255) & ~(size_t)255;
        return p;
    };
    uint16_t* hidden_bf = (uint16_t*)take(4096ull * 2048 * 2);
    uint16_t* WqkvzT    = (uint16_t*)take(12288ull * 2048 * 2);
    uint16_t* WoutT     = (uint16_t*)take(2048ull * 4096 * 2);
    float*    qkvz      = (float*)take(4096ull * 12288 * 4);
    float*    betab     = (float*)take(4096ull * 32 * 4);
    float*    gb        = (float*)take(4096ull * 32 * 4);
    uint16_t* qg        = (uint16_t*)take(4096ull * 2048 * 2);
    uint16_t* kg        = (uint16_t*)take(4096ull * 2048 * 2);
    float*    vg        = (float*)take(4096ull * 4096 * 4);
    float*    ob        = (float*)take(4096ull * 4096 * 4);
    uint16_t* xg        = (uint16_t*)take(4096ull * 4096 * 2);

    // 0/1: convert + transpose weights to bf16 [N][K]
    cvt_bf16_kernel<<<(4096 * 2048) / 256, 256, 0, stream>>>(hidden, hidden_bf,
                                                             (size_t)4096 * 2048);
    transpose_bf16_kernel<<<(12288 * 2048) / 256, 256, 0, stream>>>(W_qkvz, WqkvzT, 2048, 12288);
    transpose_bf16_kernel<<<(4096 * 2048) / 256, 256, 0, stream>>>(W_out, WoutT, 4096, 2048);

    // 2: qkvz = hidden @ W_qkvz   (M=4096, N=12288, K=2048)
    gemm_bf16_nt<<<dim3(96, 32), 256, 0, stream>>>(hidden_bf, WqkvzT, qkvz, 4096, 12288, 2048);

    // 3: ba -> beta, g
    ba_gate_kernel<<<4096, 256, 0, stream>>>(hidden, W_ba, dt_bias, A_log, betab, gb);

    // 4: conv + silu + l2norm -> q,k bf16 / v f32
    conv_silu_norm_kernel<<<4096, 256, 0, stream>>>(qkvz, conv_w, qg, kg, vg);

    // 5: chunked gated delta rule (one block per value head)
    hipFuncSetAttribute(reinterpret_cast<const void*>(delta_rule_kernel),
                        hipFuncAttributeMaxDynamicSharedMemorySize, DELTA_SMEM);
    delta_rule_kernel<<<32, 256, DELTA_SMEM, stream>>>(qg, kg, vg, gb, betab, ob);

    // 6: gating + RMSNorm -> xg bf16
    gate_norm_kernel<<<4096, 256, 0, stream>>>(ob, qkvz, norm_w, xg);

    // 7: out = xg @ W_out  (M=4096, N=2048, K=4096)
    gemm_bf16_nt<<<dim3(16, 32), 256, 0, stream>>>(xg, WoutT, out, 4096, 2048, 4096);
}
